// GraphReasoner_85040352461310
// MI455X (gfx1250) — matrix operand order repaired and checked by reading, not compiled, not run on hardware
//
#include <hip/hip_runtime.h>
#include <hip/hip_bf16.h>

// ---------------------------------------------------------------------------
// MI455X (gfx1250) GraphReasoner: B=16, N=512, D=512, R=8, STEPS=2
// Compute-bound (~190 GFLOP) -> bf16 WMMA (v_wmma_f32_16x16x32_bf16) with f32
// accumulation. Weight transpose/convert hoisted to one-time prep kernels.
// GEMM kernels are double-buffered (ping-pong LDS) with software pipelining;
// pure-copy B tiles use GLOBAL_LOAD_ASYNC_TO_LDS (+ s_wait_asynccnt) when the
// builtins exist, register-staged copies otherwise.
// ---------------------------------------------------------------------------

#define DEV __device__ __forceinline__

typedef __attribute__((ext_vector_type(16))) __bf16 v16bf;
typedef __attribute__((ext_vector_type(8)))  __bf16 v8bf;
typedef __attribute__((ext_vector_type(8)))  float  v8f;

constexpr int Bn = 16;   // batch
constexpr int Nn = 512;  // nodes
constexpr int Dn = 512;  // feature dim
constexpr int Rn = 8;    // relations

#if defined(__gfx1250__) &&                                        \
    __has_builtin(__builtin_amdgcn_global_load_async_to_lds_b128) && \
    __has_builtin(__builtin_amdgcn_global_load_async_to_lds_b64) &&  \
    __has_builtin(__builtin_amdgcn_s_wait_asynccnt)
#define HAS_ASYNC_LDS 1
#else
#define HAS_ASYNC_LDS 0
#endif

#if HAS_ASYNC_LDS
// Parameter types per hipcc diagnostic: width-matched int vectors, AS1 global
// source / AS3 LDS destination. Flat addr[31:0] == LDS byte offset (ISA 10.2
// aperture calc), so integer round-trips produce valid AS pointers.
typedef __attribute__((__vector_size__(4 * sizeof(int)))) int v4i_t;
typedef __attribute__((__vector_size__(2 * sizeof(int)))) int v2i_t;
typedef __attribute__((address_space(1))) v4i_t* g4p;
typedef __attribute__((address_space(3))) v4i_t* l4p;
typedef __attribute__((address_space(1))) v2i_t* g2p;
typedef __attribute__((address_space(3))) v2i_t* l2p;

DEV void async_b128(const void* g, void* l) {
  __builtin_amdgcn_global_load_async_to_lds_b128(
      (g4p)(unsigned long long)(size_t)g, (l4p)(unsigned)(size_t)l, 0, 0);
}
DEV void async_b64(const void* g, void* l) {
  __builtin_amdgcn_global_load_async_to_lds_b64(
      (g2p)(unsigned long long)(size_t)g, (l2p)(unsigned)(size_t)l, 0, 0);
}
#endif

// f32 -> bf16, round-to-nearest-even
DEV unsigned short f2bf(float f) {
  unsigned u = __float_as_uint(f);
  u += 0x7FFFu + ((u >> 16) & 1u);
  return (unsigned short)(u >> 16);
}
DEV unsigned pack2(float a, float b) {
  return (unsigned)f2bf(a) | ((unsigned)f2bf(b) << 16);
}

// A fragment (16x32 bf16) per ISA layout: lane L<16 -> M=L, K={0..7,16..23};
// lane L>=16 -> M=L-16, K={8..15,24..31}. Two 16-byte LDS reads per lane.
DEV v16bf ldsA_frag(const unsigned short* ldsA, int lane) {
  int m   = lane & 15;
  int klo = (lane & 16) ? 8 : 0;
  const unsigned short* row = ldsA + m * 32;
  v16bf f;
  *((v8bf*)&f)     = *(const v8bf*)(row + klo);
  *((v8bf*)&f + 1) = *(const v8bf*)(row + klo + 16);
  return f;
}

// B fragment (32x16 bf16): lane L -> N=L%16, K={0..7,16..23} (L<16) / {8..15,24..31} (L>=16).
DEV v16bf ldsB_frag(const unsigned short* ldsB, int lane, int n0) {
  const __bf16* bp = (const __bf16*)ldsB;
  int n  = n0 + (lane & 15);
  int kb = (lane & 16) ? 8 : 0;
  v16bf f;
#pragma unroll
  for (int s = 0; s < 8; ++s) {
    f[s]     = bp[(kb + s) * 128 + n];
    f[8 + s] = bp[(16 + kb + s) * 128 + n];
  }
  return f;
}

DEV v8f wmma_bf16(v16bf a, v16bf b, v8f c) {
  return __builtin_amdgcn_wmma_f32_16x16x32_bf16(
      false, a, false, b, (short)0, c, false, false);
}

// ---------------------------------------------------------------------------
__global__ __launch_bounds__(256) void bias_kernel(const float* __restrict__ msg_b,
                                                   float* __restrict__ bias_sum) {
  int e = blockIdx.x * 256 + threadIdx.x;
  float s = 0.f;
#pragma unroll
  for (int r = 0; r < Rn; ++r) s += msg_b[r * Dn + e];
  bias_sum[e] = s;
}

// ---------------------------------------------------------------------------
// One-time: dst[c][r] = bf16(src[r][c]) per z-grid matrix (LDS-tiled 32x32).
// ---------------------------------------------------------------------------
__global__ __launch_bounds__(256) void transpose_cvt(const float* __restrict__ src,
                                                     unsigned short* __restrict__ dst,
                                                     int rows, int cols) {
  const size_t msz = (size_t)rows * cols;
  src += (size_t)blockIdx.z * msz;
  dst += (size_t)blockIdx.z * msz;
  const int r0 = blockIdx.y * 32, c0 = blockIdx.x * 32;
  __shared__ unsigned short tile[32][33];
  const int t = threadIdx.x;
  {
    int rr = t >> 3, c4 = (t & 7) << 2;
    float4 v = *(const float4*)(src + (size_t)(r0 + rr) * cols + c0 + c4);
    tile[c4 + 0][rr] = f2bf(v.x);
    tile[c4 + 1][rr] = f2bf(v.y);
    tile[c4 + 2][rr] = f2bf(v.z);
    tile[c4 + 3][rr] = f2bf(v.w);
  }
  __syncthreads();
  {
    int cc = t >> 3, r4 = (t & 7) << 2;
    uint2 p;
    p.x = (unsigned)tile[cc][r4 + 0] | ((unsigned)tile[cc][r4 + 1] << 16);
    p.y = (unsigned)tile[cc][r4 + 2] | ((unsigned)tile[cc][r4 + 3] << 16);
    *(uint2*)(dst + (size_t)(c0 + cc) * rows + r0 + r4) = p;
  }
}

// ---------------------------------------------------------------------------
__global__ __launch_bounds__(256) void cvt_kernel(const float* __restrict__ src,
                                                  unsigned short* __restrict__ dst) {
  int i = blockIdx.x * 256 + threadIdx.x;  // float4 units
  float4 v = ((const float4*)src)[i];
  uint2 p; p.x = pack2(v.x, v.y); p.y = pack2(v.z, v.w);
  ((uint2*)dst)[i] = p;
}

// ---------------------------------------------------------------------------
// HW[b,r] = Hbf[b] (NxD) @ Wt[r] ([k][n] bf16), output bf16.
// 32x128 tile, K-step 32, double-buffered; both tiles are pure copies.
// ---------------------------------------------------------------------------
__global__ __launch_bounds__(256) void hw_kernel(const unsigned short* __restrict__ hbf,
                                                 const unsigned short* __restrict__ wt,
                                                 unsigned short* __restrict__ hw) {
  __shared__ alignas(32) unsigned short As[2][32 * 32];
  __shared__ alignas(32) unsigned short Bs[2][32 * 128];
  int t = blockIdx.x;
  const int nT = t & 3;  t >>= 2;
  const int mT = t & 15; t >>= 4;
  const int r  = t & 7;  t >>= 3;
  const int b  = t;
  const int i0 = mT * 32, e0 = nT * 128;
  const int tid = threadIdx.x, lane = tid & 31, w = tid >> 5;
  const int msub = w >> 2, nb = (w & 3) * 32;
  v8f acc0 = {}, acc1 = {};

  const int row = tid >> 3, c4 = (tid & 7) << 2;   // A coords (32x32, uint2/thread)
  const int cs  = (tid & 7) << 4;                  // B coords (32x128, 2xuint4/thread)
  const unsigned short* aSrc = hbf + ((size_t)b * Nn + i0 + row) * Dn + c4;   // +k0
  const unsigned short* bSrc = wt + ((size_t)r * Dn + row) * Dn + e0 + cs;    // +k0*Dn
  unsigned short* aD[2] = { &As[0][row * 32 + c4], &As[1][row * 32 + c4] };
  unsigned short* bD[2] = { &Bs[0][row * 128 + cs], &Bs[1][row * 128 + cs] };

  auto compute = [&](int p) {
    v16bf a = ldsA_frag(&As[p][msub * 16 * 32], lane);
    acc0 = wmma_bf16(a, ldsB_frag(&Bs[p][0], lane, nb), acc0);
    acc1 = wmma_bf16(a, ldsB_frag(&Bs[p][0], lane, nb + 16), acc1);
  };

#if HAS_ASYNC_LDS
  auto issue = [&](int k0, int p) {
    async_b64(aSrc + k0, aD[p]);
    async_b128(bSrc + (size_t)k0 * Dn, bD[p]);
    async_b128(bSrc + (size_t)k0 * Dn + 8, bD[p] + 8);
  };
  issue(0, 0);
  int p = 0;
  for (int k0 = 0; k0 < Dn - 32; k0 += 32, p ^= 1) {
    issue(k0 + 32, p ^ 1);
    __builtin_amdgcn_s_wait_asynccnt(3);
    __syncthreads();
    compute(p);
    __syncthreads();
  }
  __builtin_amdgcn_s_wait_asynccnt(0);
  __syncthreads();
  compute(p);
#else
  uint2 ar = *(const uint2*)aSrc;
  uint4 br0 = ((const uint4*)bSrc)[0];
  uint4 br1 = ((const uint4*)bSrc)[1];
  int p = 0;
  for (int k0 = 0; k0 < Dn; k0 += 32, p ^= 1) {
    *(uint2*)aD[p] = ar;
    ((uint4*)bD[p])[0] = br0;
    ((uint4*)bD[p])[1] = br1;
    if (k0 + 32 < Dn) {
      ar  = *(const uint2*)(aSrc + k0 + 32);
      br0 = ((const uint4*)(bSrc + (size_t)(k0 + 32) * Dn))[0];
      br1 = ((const uint4*)(bSrc + (size_t)(k0 + 32) * Dn))[1];
    }
    __syncthreads();
    compute(p);
    __syncthreads();
  }
#endif

  int m = msub * 16 + ((lane >> 4) << 3);
  int n = lane & 15;
  unsigned short* out = hw + (((size_t)b * Rn + r) * Nn + i0 + m) * Dn + e0 + nb + n;
#pragma unroll
  for (int v = 0; v < 8; ++v) {
    out[(size_t)v * Dn]      = f2bf(acc0[v]);
    out[(size_t)v * Dn + 16] = f2bf(acc1[v]);
  }
}

// ---------------------------------------------------------------------------
// aggbf[b] = bf16( sum_r E[b,r] @ HW[b,r] + bias ).  Flattened (r,k) pipeline:
// A = edges f32->bf16 via register staging, B = HW copies (async when possible).
// ---------------------------------------------------------------------------
__global__ __launch_bounds__(256) void agg_kernel(const float* __restrict__ edges,
                                                  const unsigned short* __restrict__ hw,
                                                  const float* __restrict__ bias,
                                                  unsigned short* __restrict__ aggbf) {
  __shared__ alignas(32) unsigned short As[2][32 * 32];
  __shared__ alignas(32) unsigned short Bs[2][32 * 128];
  int t = blockIdx.x;
  const int nT = t & 3;  t >>= 2;
  const int mT = t & 15; t >>= 4;
  const int b  = t;
  const int i0 = mT * 32, e0 = nT * 128;
  const int tid = threadIdx.x, lane = tid & 31, w = tid >> 5;
  const int msub = w >> 2, nb = (w & 3) * 32;
  v8f acc0 = {}, acc1 = {};

  const int row = tid >> 3, c4 = (tid & 7) << 2;
  const int cs  = (tid & 7) << 4;
  unsigned short* aD[2] = { &As[0][row * 32 + c4], &As[1][row * 32 + c4] };
  unsigned short* bD[2] = { &Bs[0][row * 128 + cs], &Bs[1][row * 128 + cs] };

  constexpr int TOT = Rn * (Nn / 32);  // 128 tiles
  auto aPtr = [&](int q) {
    int r = q >> 4, k0 = (q & 15) << 5;
    return edges + (((size_t)b * Rn + r) * Nn + i0 + row) * Nn + k0 + c4;
  };
  auto bPtr = [&](int q) {
    int r = q >> 4, k0 = (q & 15) << 5;
    return hw + (((size_t)b * Rn + r) * Nn + k0 + row) * Dn + e0 + cs;
  };
  auto compute = [&](int p) {
    v16bf a = ldsA_frag(&As[p][msub * 16 * 32], lane);
    acc0 = wmma_bf16(a, ldsB_frag(&Bs[p][0], lane, nb), acc0);
    acc1 = wmma_bf16(a, ldsB_frag(&Bs[p][0], lane, nb + 16), acc1);
  };
  auto stageA = [&](float4 v, int p) {
    uint2 pk; pk.x = pack2(v.x, v.y); pk.y = pack2(v.z, v.w);
    *(uint2*)aD[p] = pk;
  };

#if HAS_ASYNC_LDS
  auto issueB = [&](int q, int p) {
    const unsigned short* s = bPtr(q);
    async_b128(s, bD[p]);
    async_b128(s + 8, bD[p] + 8);
  };
  float4 ar = *(const float4*)aPtr(0);
  issueB(0, 0);
  int p = 0;
  for (int q = 0; q < TOT - 1; ++q, p ^= 1) {
    stageA(ar, p);
    ar = *(const float4*)aPtr(q + 1);
    if (q + 2 < TOT) __builtin_prefetch(aPtr(q + 2), 0, 0);  // global_prefetch_b8
    issueB(q + 1, p ^ 1);
    __builtin_amdgcn_s_wait_asynccnt(2);
    __syncthreads();
    compute(p);
    __syncthreads();
  }
  stageA(ar, p);
  __builtin_amdgcn_s_wait_asynccnt(0);
  __syncthreads();
  compute(p);
#else
  float4 ar = *(const float4*)aPtr(0);
  uint4 br0 = ((const uint4*)bPtr(0))[0];
  uint4 br1 = ((const uint4*)bPtr(0))[1];
  int p = 0;
  for (int q = 0; q < TOT; ++q, p ^= 1) {
    stageA(ar, p);
    ((uint4*)bD[p])[0] = br0;
    ((uint4*)bD[p])[1] = br1;
    if (q + 1 < TOT) {
      ar  = *(const float4*)aPtr(q + 1);
      br0 = ((const uint4*)bPtr(q + 1))[0];
      br1 = ((const uint4*)bPtr(q + 1))[1];
      if (q + 2 < TOT) __builtin_prefetch(aPtr(q + 2), 0, 0);
    }
    __syncthreads();
    compute(p);
    __syncthreads();
  }
#endif

  int m = msub * 16 + ((lane >> 4) << 3);
  int n = lane & 15;
  float bi0 = bias[e0 + nb + n], bi1 = bias[e0 + nb + 16 + n];
  unsigned short* out = aggbf + ((size_t)b * Nn + i0 + m) * Dn + e0 + nb + n;
#pragma unroll
  for (int v = 0; v < 8; ++v) {
    out[(size_t)v * Dn]      = f2bf(acc0[v] + bi0);
    out[(size_t)v * Dn + 16] = f2bf(acc1[v] + bi1);
  }
}

// ---------------------------------------------------------------------------
// Fused GRU: gi = aggbf @ wihT + b_ih ; gh = hbf @ whhT + b_hh ; gates + blend.
// 16x128 tile; wave w owns E-cols w*16..+15 with 6 gate accumulators.
// ---------------------------------------------------------------------------
__global__ __launch_bounds__(256) void gru_kernel(const unsigned short* __restrict__ aggbf,
                                                  const unsigned short* __restrict__ hbf,
                                                  const float* __restrict__ hf32,
                                                  const unsigned short* __restrict__ wihT,
                                                  const unsigned short* __restrict__ whhT,
                                                  const float* __restrict__ bih,
                                                  const float* __restrict__ bhh,
                                                  float* __restrict__ hout) {
  __shared__ alignas(32) unsigned short Aag[16 * 32];
  __shared__ alignas(32) unsigned short Ah[16 * 32];
  __shared__ alignas(32) unsigned short Bih[3][32 * 128];
  __shared__ alignas(32) unsigned short Bhh[3][32 * 128];
  int t = blockIdx.x;
  const int nT = t & 3;  t >>= 2;
  const int mT = t & 31; t >>= 5;
  const int b  = t;
  const int i0 = mT * 16, e0 = nT * 128;
  const int tid = threadIdx.x, lane = tid & 31, w = tid >> 5;
  const int n0 = w * 16;
  v8f gi0 = {}, gi1 = {}, gi2 = {}, gh0 = {}, gh1 = {}, gh2 = {};
  const unsigned short* aggB = aggbf + ((size_t)b * Nn + i0) * Dn;
  const unsigned short* hB   = hbf + ((size_t)b * Nn + i0) * Dn;

  for (int k0 = 0; k0 < Dn; k0 += 32) {
    {
      int tt = tid & 127;
      int row = tt >> 3, c4 = (tt & 7) << 2;
      const unsigned short* src = (tid < 128 ? aggB : hB) + (size_t)row * Dn + k0 + c4;
      unsigned short* dst = (tid < 128 ? Aag : Ah) + row * 32 + c4;
      *(uint2*)dst = *(const uint2*)src;
    }
    {
      int k = tid >> 3, cs = (tid & 7) << 4;
      const size_t rowoff = (size_t)(k0 + k) * (3 * Dn) + e0 + cs;
#pragma unroll
      for (int g = 0; g < 3; ++g) {
        const uint4* s1 = (const uint4*)(wihT + rowoff + g * Dn);
        const uint4* s2 = (const uint4*)(whhT + rowoff + g * Dn);
        uint4* d1 = (uint4*)(&Bih[g][k * 128 + cs]);
        uint4* d2 = (uint4*)(&Bhh[g][k * 128 + cs]);
        d1[0] = s1[0]; d1[1] = s1[1];
        d2[0] = s2[0]; d2[1] = s2[1];
      }
    }
    __syncthreads();
    v16bf aA = ldsA_frag(Aag, lane);
    v16bf aH = ldsA_frag(Ah, lane);
    gi0 = wmma_bf16(aA, ldsB_frag(Bih[0], lane, n0), gi0);
    gi1 = wmma_bf16(aA, ldsB_frag(Bih[1], lane, n0), gi1);
    gi2 = wmma_bf16(aA, ldsB_frag(Bih[2], lane, n0), gi2);
    gh0 = wmma_bf16(aH, ldsB_frag(Bhh[0], lane, n0), gh0);
    gh1 = wmma_bf16(aH, ldsB_frag(Bhh[1], lane, n0), gh1);
    gh2 = wmma_bf16(aH, ldsB_frag(Bhh[2], lane, n0), gh2);
    __syncthreads();
  }
  int m  = (lane >> 4) << 3;
  int nl = lane & 15;
  int e  = e0 + n0 + nl;
  float bi_r = bih[e], bi_z = bih[Dn + e], bi_n = bih[2 * Dn + e];
  float bh_r = bhh[e], bh_z = bhh[Dn + e], bh_n = bhh[2 * Dn + e];
  const float* hrow = hf32 + ((size_t)b * Nn + i0 + m) * Dn + e;
  float* orow = hout + ((size_t)b * Nn + i0 + m) * Dn + e;
#pragma unroll
  for (int v = 0; v < 8; ++v) {
    float ir = gi0[v] + bi_r, iz = gi1[v] + bi_z, inn = gi2[v] + bi_n;
    float hr = gh0[v] + bh_r, hz = gh1[v] + bh_z, hn = gh2[v] + bh_n;
    float rr = 1.f / (1.f + __expf(-(ir + hr)));
    float zz = 1.f / (1.f + __expf(-(iz + hz)));
    float nt = tanhf(inn + rr * hn);
    float ho = hrow[(size_t)v * Dn];
    orow[(size_t)v * Dn] = (1.f - zz) * nt + zz * ho;
  }
}

// ---------------------------------------------------------------------------
__global__ __launch_bounds__(128) void mean_kernel(const float* __restrict__ h,
                                                   float* __restrict__ out) {
  int b = blockIdx.x >> 2;
  int e = ((blockIdx.x & 3) << 7) + threadIdx.x;
  float s = 0.f;
  for (int n = 0; n < Nn; ++n) s += h[((size_t)b * Nn + n) * Dn + e];
  out[b * Dn + e] = s * (1.f / (float)Nn);
}

// ---------------------------------------------------------------------------
extern "C" void kernel_launch(void* const* d_in, const int* in_sizes, int n_in,
                              void* d_out, int out_size, void* d_ws, size_t ws_size,
                              hipStream_t stream) {
  (void)in_sizes; (void)n_in; (void)out_size; (void)ws_size;
  const float* nodes = (const float*)d_in[0];  // [B,N,D]
  const float* edges = (const float*)d_in[1];  // [B,R,N,N]
  const float* msg_W = (const float*)d_in[2];  // [R,D,D]
  const float* msg_b = (const float*)d_in[3];  // [R,D]
  const float* w_ih  = (const float*)d_in[4];  // [3D,D]
  const float* w_hh  = (const float*)d_in[5];  // [3D,D]
  const float* b_ih  = (const float*)d_in[6];  // [3D]
  const float* b_hh  = (const float*)d_in[7];  // [3D]
  float* out = (float*)d_out;                  // [B,D]

  // Workspace layout (~125 MB), all offsets 4KB-aligned
  char* ws = (char*)d_ws;
  size_t off = 0;
  float* bias_sum = (float*)(ws + off);           off += 4096;
  unsigned short* aggbf = (unsigned short*)(ws + off);
  off += (size_t)Bn * Nn * Dn * 2;                // 8.4 MB
  unsigned short* HW = (unsigned short*)(ws + off);
  off += (size_t)Bn * Rn * Nn * Dn * 2;           // 67 MB
  float* hA = (float*)(ws + off);
  off += (size_t)Bn * Nn * Dn * 4;                // 16.8 MB
  float* hB = (float*)(ws + off);
  off += (size_t)Bn * Nn * Dn * 4;                // 16.8 MB
  unsigned short* hbf = (unsigned short*)(ws + off);
  off += (size_t)Bn * Nn * Dn * 2;                // 8.4 MB
  unsigned short* Wt = (unsigned short*)(ws + off);
  off += (size_t)Rn * Dn * Dn * 2;                // 4.2 MB
  unsigned short* wihT = (unsigned short*)(ws + off);
  off += (size_t)3 * Dn * Dn * 2;                 // 1.6 MB
  unsigned short* whhT = (unsigned short*)(ws + off);
  off += (size_t)3 * Dn * Dn * 2;                 // 1.6 MB

  const int hwGrid  = Bn * Rn * (Nn / 32) * (Dn / 128);  // 8192
  const int aggGrid = Bn * (Nn / 32) * (Dn / 128);       // 1024
  const int gruGrid = Bn * (Nn / 16) * (Dn / 128);       // 2048
  const int cvtGrid = (Bn * Nn * Dn / 4) / 256;          // 4096

  // one-time prep
  bias_kernel<<<Dn / 256, 256, 0, stream>>>(msg_b, bias_sum);
  transpose_cvt<<<dim3(Dn / 32, Dn / 32, Rn), 256, 0, stream>>>(msg_W, Wt, Dn, Dn);
  transpose_cvt<<<dim3(Dn / 32, 3 * Dn / 32, 1), 256, 0, stream>>>(w_ih, wihT, 3 * Dn, Dn);
  transpose_cvt<<<dim3(Dn / 32, 3 * Dn / 32, 1), 256, 0, stream>>>(w_hh, whhT, 3 * Dn, Dn);

  // step 0: h = nodes -> hA
  cvt_kernel<<<cvtGrid, 256, 0, stream>>>(nodes, hbf);
  hw_kernel<<<hwGrid, 256, 0, stream>>>(hbf, Wt, HW);
  agg_kernel<<<aggGrid, 256, 0, stream>>>(edges, HW, bias_sum, aggbf);
  gru_kernel<<<gruGrid, 256, 0, stream>>>(aggbf, hbf, nodes, wihT, whhT, b_ih, b_hh, hA);

  // step 1: h = hA -> hB
  cvt_kernel<<<cvtGrid, 256, 0, stream>>>(hA, hbf);
  hw_kernel<<<hwGrid, 256, 0, stream>>>(hbf, Wt, HW);
  agg_kernel<<<aggGrid, 256, 0, stream>>>(edges, HW, bias_sum, aggbf);
  gru_kernel<<<gruGrid, 256, 0, stream>>>(aggbf, hbf, hA, wihT, whhT, b_ih, b_hh, hB);

  mean_kernel<<<Bn * (Dn / 128), 128, 0, stream>>>(hB, out);
}
